// ModelR_37022618091886
// MI455X (gfx1250) — matrix operand order repaired and checked by reading, not compiled, not run on hardware
//
#include <hip/hip_runtime.h>

typedef __attribute__((ext_vector_type(16))) _Float16 v16h;
typedef __attribute__((ext_vector_type(8)))  float    v8f;
typedef __attribute__((ext_vector_type(4)))  float    f32x4;
typedef __attribute__((ext_vector_type(4)))  unsigned int u32x4;
typedef __attribute__((ext_vector_type(2)))  unsigned int u32x2;

#define BM 128
#define BN 128
#define BK 32
#define LDH 48   // LDS row stride in halves: 96 bytes = 6*16B -> b128-aligned frag reads

union Frag { u32x4 u[2]; v16h v; };

// C = act(A @ B + bias). A: f32 (first layer) or f16 (hidden). B: f32 row-major [K][N].
// Output: f16 with leaky-relu (hidden) or f32 raw (last layer).
// Block 128x128, 8 waves in a 4x2 grid, each wave owns a 32x64 tile = 2x4 WMMA accums.
template<bool A_F32, bool ACT>
__global__ __launch_bounds__(256)
void gemm_k(const void* __restrict__ Ap, const float* __restrict__ Bw,
            const float* __restrict__ bias, void* __restrict__ Cp,
            int M, int N, int K)
{
  __shared__ _Float16 lsA[2][BM * LDH];
  __shared__ _Float16 lsB[2][BN * LDH];   // stored transposed: [n][k]

  const int tid  = threadIdx.x;
  const int m0   = blockIdx.y * BM;
  const int n0   = blockIdx.x * BN;
  const int lane = tid & 31;
  const int wave = tid >> 5;
  const int wm   = wave >> 1;     // 0..3 -> rows wm*32
  const int wn   = wave & 1;      // 0..1 -> cols wn*64
  const int r    = lane & 15;
  const int hi   = lane >> 4;     // which half of the wave

  f32x4 stA[4];
  u32x4 stAh[2];
  f32x4 stB[4];

  const int KT = K / BK;

  auto loadG = [&](int kt) {
    const int k0 = kt * BK;
    if constexpr (A_F32) {
      const float* A = (const float*)Ap;
      const int kc = (tid & 7) * 4;
      const int rr = tid >> 3;                    // 0..31
#pragma unroll
      for (int p = 0; p < 4; ++p)
        stA[p] = *(const f32x4*)(A + (size_t)(m0 + p*32 + rr) * K + k0 + kc);
    } else {
      const _Float16* A = (const _Float16*)Ap;
      const int kc = (tid & 3) * 8;
      const int rr = tid >> 2;                    // 0..63
#pragma unroll
      for (int p = 0; p < 2; ++p)
        stAh[p] = *(const u32x4*)(A + (size_t)(m0 + p*64 + rr) * K + k0 + kc);
    }
    // B tile: 32 k-rows x 128 cols of f32
    const int nc = (tid & 31) * 4;                // 0..124
    const int kr = tid >> 5;                      // 0..7
#pragma unroll
    for (int p = 0; p < 4; ++p) {
      const int gn = n0 + nc;
      f32x4 v = {0.f, 0.f, 0.f, 0.f};
      if (gn < N)                                 // N % 4 == 0 -> all-or-nothing
        v = *(const f32x4*)(Bw + (size_t)(k0 + p*8 + kr) * N + gn);
      stB[p] = v;
    }
  };

  auto storeT = [&](int buf) {
    if constexpr (A_F32) {
      const int kc = (tid & 7) * 4;
      const int rr = tid >> 3;
#pragma unroll
      for (int p = 0; p < 4; ++p) {
        union { _Float16 h[4]; u32x2 u; } pk;
        pk.h[0] = (_Float16)stA[p].x; pk.h[1] = (_Float16)stA[p].y;
        pk.h[2] = (_Float16)stA[p].z; pk.h[3] = (_Float16)stA[p].w;
        *(u32x2*)&lsA[buf][(p*32 + rr) * LDH + kc] = pk.u;
      }
    } else {
      const int kc = (tid & 3) * 8;
      const int rr = tid >> 2;
#pragma unroll
      for (int p = 0; p < 2; ++p)
        *(u32x4*)&lsA[buf][(p*64 + rr) * LDH + kc] = stAh[p];
    }
    const int nc = (tid & 31) * 4;
    const int kr = tid >> 5;
#pragma unroll
    for (int p = 0; p < 4; ++p) {
      lsB[buf][(nc+0)*LDH + p*8 + kr] = (_Float16)stB[p].x;
      lsB[buf][(nc+1)*LDH + p*8 + kr] = (_Float16)stB[p].y;
      lsB[buf][(nc+2)*LDH + p*8 + kr] = (_Float16)stB[p].z;
      lsB[buf][(nc+3)*LDH + p*8 + kr] = (_Float16)stB[p].w;
    }
  };

  v8f acc[2][4];
  const v8f vzero = {0.f,0.f,0.f,0.f,0.f,0.f,0.f,0.f};
#pragma unroll
  for (int i = 0; i < 2; ++i)
#pragma unroll
    for (int j = 0; j < 4; ++j)
      acc[i][j] = vzero;

  loadG(0);
  storeT(0);
  __syncthreads();

  for (int kt = 0; kt < KT; ++kt) {
    const int cur = kt & 1;
    if (kt + 1 < KT) loadG(kt + 1);

    // Fragment loads per ISA 7.12.2 16-bit layouts:
    // A elem i -> K = i + (i>=8?8:0) + (lane>=16?8:0)  => two b128 at k {hi*8, 16+hi*8}
    // B elem i -> K = i + (i>=8?8:0) + (lane>=16?8:0), col = lane%16 => two b128 at k {hi*8, 16+hi*8} (transposed LDS)
    Frag af[2], bfr[4];
#pragma unroll
    for (int tm = 0; tm < 2; ++tm) {
      const int row = wm*32 + tm*16 + r;
      af[tm].u[0] = *(const u32x4*)&lsA[cur][row*LDH + hi*8];
      af[tm].u[1] = *(const u32x4*)&lsA[cur][row*LDH + 16 + hi*8];
    }
#pragma unroll
    for (int tn = 0; tn < 4; ++tn) {
      const int col = wn*64 + tn*16 + r;
      bfr[tn].u[0] = *(const u32x4*)&lsB[cur][col*LDH + hi*8];
      bfr[tn].u[1] = *(const u32x4*)&lsB[cur][col*LDH + 16 + hi*8];
    }
#pragma unroll
    for (int tm = 0; tm < 2; ++tm)
#pragma unroll
      for (int tn = 0; tn < 4; ++tn)
        acc[tm][tn] = __builtin_amdgcn_wmma_f32_16x16x32_f16(
            false, af[tm].v, false, bfr[tn].v, (short)0, acc[tm][tn], false, false);

    if (kt + 1 < KT) storeT(cur ^ 1);
    __syncthreads();
  }

  // Epilogue. C/D layout: VGPR v holds row = v + (lane>=16?8:0), col = lane%16.
#pragma unroll
  for (int tm = 0; tm < 2; ++tm) {
    const int grow = m0 + wm*32 + tm*16 + hi*8;
#pragma unroll
    for (int tn = 0; tn < 4; ++tn) {
      const int gc = n0 + wn*64 + tn*16 + r;
      if (gc < N) {
        const float bv = bias[gc];
#pragma unroll
        for (int v = 0; v < 8; ++v) {
          float f = acc[tm][tn][v] + bv;
          if constexpr (ACT) {
            f = f > 0.f ? f : 0.01f * f;
            ((_Float16*)Cp)[(size_t)(grow + v) * N + gc] = (_Float16)f;
          } else {
            ((float*)Cp)[(size_t)(grow + v) * N + gc] = f;
          }
        }
      }
    }
  }
}

// Scatter y[8192][2080] -> out[64][64][8192] (upper triangle incl. diag; diag abs'ed).
// LDS transpose so reads are coalesced along l and writes coalesced along b.
__global__ __launch_bounds__(256)
void scatter_k(const float* __restrict__ y, float* __restrict__ out)
{
  __shared__ float tile[64][65];
  const int tid = threadIdx.x;
  const int l0  = blockIdx.x * 64;
  const int b0  = blockIdx.y * 64;

  const int cl = (tid & 15) * 4;
  const int rb = tid >> 4;                        // 0..15
#pragma unroll
  for (int p = 0; p < 4; ++p) {
    const int bb = p*16 + rb;
    f32x4 v = {0.f, 0.f, 0.f, 0.f};
    if (l0 + cl < 2080)                           // 2080 % 4 == 0 -> all-or-nothing
      v = *(const f32x4*)(y + (size_t)(b0 + bb) * 2080 + l0 + cl);
    tile[bb][cl+0] = v.x; tile[bb][cl+1] = v.y;
    tile[bb][cl+2] = v.z; tile[bb][cl+3] = v.w;
  }
  __syncthreads();

  const int lt = tid >> 2;                        // 0..63
  const int bq = (tid & 3) * 16;                  // 0,16,32,48
  const int l  = l0 + lt;
  if (l < 2080) {
    // invert l -> (row,col): off(row) = row*(129-row)/2; col = 63 - (l - off(row))
    int row = (int)(64.5f - sqrtf(64.5f*64.5f - 2.0f*(float)l));
    if (row < 0) row = 0;
    if (row > 63) row = 63;
    while (row > 0 && row*(129 - row)/2 > l) --row;
    while (row < 63 && (row+1)*(129 - (row+1))/2 <= l) ++row;
    const int col  = 63 - (l - row*(129 - row)/2);
    const bool dia = (row == col);
    const size_t obase = (size_t)(row*64 + col) * 8192 + b0 + bq;
#pragma unroll
    for (int j = 0; j < 16; ++j) {
      float v = tile[bq + j][lt];
      if (dia && v <= 0.f) v = -v;
      out[obase + j] = v;
    }
  }
}

// Zero the strictly-lower triangle of the output (never touched by scatter_k).
__global__ __launch_bounds__(256)
void zlow_k(float* __restrict__ out)
{
  const int rc = blockIdx.y;
  const int rr = rc >> 6, cc = rc & 63;
  if (cc >= rr) return;
  const size_t i = (size_t)blockIdx.x * 256 + threadIdx.x;   // f32x4 index, 2048 per (r,c)
  const f32x4 z = {0.f, 0.f, 0.f, 0.f};
  ((f32x4*)out)[(size_t)rc * 2048 + i] = z;
}

extern "C" void kernel_launch(void* const* d_in, const int* in_sizes, int n_in,
                              void* d_out, int out_size, void* d_ws, size_t ws_size,
                              hipStream_t stream) {
  const float* x   = (const float*)d_in[0];
  const float* W1  = (const float*)d_in[1];
  const float* b1  = (const float*)d_in[2];
  const float* W2  = (const float*)d_in[3];
  const float* b2  = (const float*)d_in[4];
  const float* W21 = (const float*)d_in[5];
  const float* b21 = (const float*)d_in[6];
  const float* W22 = (const float*)d_in[7];
  const float* b22 = (const float*)d_in[8];
  const float* W3  = (const float*)d_in[9];
  const float* b3  = (const float*)d_in[10];
  float* out = (float*)d_out;

  char* ws = (char*)d_ws;
  _Float16* hA = (_Float16*)ws;                          // 8192*2048 f16 = 32 MiB
  _Float16* hB = (_Float16*)(ws + (size_t)33554432);     // 32 MiB
  float*    y  = (float*)(ws + (size_t)67108864);        // 8192*2080 f32 = 65 MiB

  const dim3 blk(256);
  const int M = 8192;

  gemm_k<true,  true ><<<dim3(2048/BN, M/BM), blk, 0, stream>>>(x,  W1,  b1,  hA, M, 2048, 1024);
  gemm_k<false, true ><<<dim3(2048/BN, M/BM), blk, 0, stream>>>(hA, W2,  b2,  hB, M, 2048, 2048);
  gemm_k<false, true ><<<dim3(2048/BN, M/BM), blk, 0, stream>>>(hB, W21, b21, hA, M, 2048, 2048);
  gemm_k<false, true ><<<dim3(2048/BN, M/BM), blk, 0, stream>>>(hA, W22, b22, hB, M, 2048, 2048);
  gemm_k<false, false><<<dim3(17,      M/BM), blk, 0, stream>>>(hB, W3,  b3,  y,  M, 2080, 2048);

  zlow_k  <<<dim3(8, 4096), blk, 0, stream>>>(out);
  scatter_k<<<dim3(33, 128), blk, 0, stream>>>(y, out);
}
